// Double_Attention_Model_81157702025531
// MI455X (gfx1250) — matrix operand order repaired and checked by reading, not compiled, not run on hardware
//
#include <hip/hip_runtime.h>
#include <hip/hip_bf16.h>

// Problem constants
#define BB 64
#define SS 32
#define NN 64
#define EE 128
#define HH 4
#define DD 32

typedef __attribute__((ext_vector_type(16))) __bf16 v16bf;
typedef __attribute__((ext_vector_type(8)))  __bf16 v8bf;
typedef __attribute__((ext_vector_type(8)))  float  v8f;
typedef __attribute__((ext_vector_type(4)))  unsigned int v4u;
typedef __attribute__((ext_vector_type(8)))  int v8i;
typedef __attribute__((ext_vector_type(4)))  int v4i;

// ISA 16-bit A-matrix (16x32) layout: element e of the per-lane fragment is
// two contiguous 8-element runs: K = kb+e (e<8), 16+kb+(e-8) (e>=8), kb=8*(lane>=16).
static __device__ __forceinline__ v16bf cat8(v8bf lo, v8bf hi) {
    return __builtin_shufflevector(lo, hi, 0, 1, 2, 3, 4, 5, 6, 7,
                                   8, 9, 10, 11, 12, 13, 14, 15);
}
static __device__ __forceinline__ v16bf load_v16bf(const __bf16* p) {
    return *(const v16bf*)p;
}
// Load an A fragment from a row pointer (row-major, contiguous K) at K-offset k0.
static __device__ __forceinline__ v16bf load_a_frag(const __bf16* rowp, int lane, int k0) {
    int kb = (lane < 16) ? 0 : 8;
    v8bf lo = *(const v8bf*)(rowp + k0 + kb);
    v8bf hi = *(const v8bf*)(rowp + k0 + 16 + kb);
    return cat8(lo, hi);
}
static __device__ __forceinline__ v8f wmma_bf16(v16bf a, v16bf b, v8f c) {
    return __builtin_amdgcn_wmma_f32_16x16x32_bf16(false, a, false, b, (short)0, c,
                                                   false, false);
}

// Tensor Data Mover: DMA one contiguous 4 KB (2048 x bf16) tile global -> LDS.
// D# group0: count=1, lds_addr, global_addr[56:0], type=2 ("image").
// D# group1: data_size=2B, 1-D tile of 2048 elements, dim0 stride 2048.
static __device__ __forceinline__ void tdm_load_tile_4kb(const __bf16* gsrc, __bf16* ldst) {
    unsigned lds_off = (unsigned)(unsigned long long)(const void*)ldst;
    unsigned long long ga = (unsigned long long)(const void*)gsrc;
    v4u g0 = {1u, lds_off, (unsigned)ga,
              (unsigned)((ga >> 32) & 0x01FFFFFFull) | 0x80000000u};
    v8i g1 = {(int)(1u << 16),            // data_size = 1 (2 bytes)
              (int)(2048u << 16),         // tensor_dim0[15:0] in bits[63:48]
              0,                          // tensor_dim0[31:16]=0, tensor_dim1 lo=0
              (int)(2048u << 16),         // tile_dim0 = 2048 in bits[127:112]
              0,                          // tile_dim1/tile_dim2 unused
              2048,                       // tensor_dim0_stride lo
              0, 0};
    v4i gz = {0, 0, 0, 0};
#if __clang_major__ >= 23
    v8i gz8 = {0, 0, 0, 0, 0, 0, 0, 0};
    __builtin_amdgcn_tensor_load_to_lds(g0, g1, gz, gz, gz8, 0);
#else
    __builtin_amdgcn_tensor_load_to_lds(g0, g1, gz, gz, 0);
#endif
    __builtin_amdgcn_s_wait_tensorcnt(0);
}

// ---------------------------------------------------------------------------
// Kernel 0: convert the 8 weight matrices (128x128 fp32, row-major W[n][k]) to
// bf16 W[n][k] so B-fragments (lane = n) are contiguous in k.
// ---------------------------------------------------------------------------
__global__ void wconv_kernel(const float* __restrict__ w0, const float* __restrict__ w1,
                             const float* __restrict__ w2, const float* __restrict__ w3,
                             const float* __restrict__ w4, const float* __restrict__ w5,
                             const float* __restrict__ w6, const float* __restrict__ w7,
                             __bf16* __restrict__ out) {
    const float* ws[8] = {w0, w1, w2, w3, w4, w5, w6, w7};
    int idx = blockIdx.x * blockDim.x + threadIdx.x;     // 8*16384 total
    int mat = idx >> 14;
    int ele = idx & 16383;                               // ele = k*128 + n
    int k = ele >> 7, n = ele & 127;
    out[mat * 16384 + n * 128 + k] = (__bf16)ws[mat][n * 128 + k];
}

// ---------------------------------------------------------------------------
// Kernel 1: projection GEMM  Out = A[M x 128](fp32) @ W^T + bias  (bf16 out)
// WT is the bf16 weight, row-major (WT[n][k]).
// Row remap (shift/mask, divisor always a power of two):
//   offset(r) = (r >> innerShift)*outerStride + (r & ((1<<innerShift)-1))*128.
// MODE 0: row-major Out[row*128 + col]          (qt, qs)
// MODE 1: kt2[b][m][h][s][d]  row=(b*S+s)*N+m   (attention-1 K layout)
// MODE 2: vt2[b][m][h][d][s]                     (attention-1 V layout)
// ---------------------------------------------------------------------------
template <int MODE>
__global__ void proj_gemm_kernel(const float* __restrict__ A,
                                 const __bf16* __restrict__ WT,
                                 const float* __restrict__ bias,
                                 __bf16* __restrict__ Out,
                                 int innerShift, long outerStride) {
    __shared__ __bf16 sA[16 * 128];
    const int tid = threadIdx.x;
    const long m0 = (long)blockIdx.x * 16;
    const long innerMask = (1L << innerShift) - 1;

    // Stage + downconvert the 16x128 fp32 A tile into LDS (8 floats/thread).
    {
        int r = tid >> 4;             // 0..15
        int c = (tid & 15) * 8;       // 0..120
        long row = m0 + r;
        long off = (row >> innerShift) * outerStride + (row & innerMask) * 128 + c;
        const float4* src = (const float4*)(A + off);
        float4 f0 = src[0], f1 = src[1];
        v8bf d;
        d[0] = (__bf16)f0.x; d[1] = (__bf16)f0.y; d[2] = (__bf16)f0.z; d[3] = (__bf16)f0.w;
        d[4] = (__bf16)f1.x; d[5] = (__bf16)f1.y; d[6] = (__bf16)f1.z; d[7] = (__bf16)f1.w;
        *(v8bf*)(sA + r * 128 + c) = d;
    }
    // Prefetch next block's A tile start (speculative).
    {
        long row = m0 + 16;
        long off = (row >> innerShift) * outerStride + (row & innerMask) * 128;
        __builtin_prefetch(&A[off], 0, 1);
    }
    __syncthreads();

    const int wave = tid >> 5, lane = tid & 31;
    const int n0 = wave * 16;
    const __bf16* arow = sA + (lane & 15) * 128;

    v8f acc = {0.f, 0.f, 0.f, 0.f, 0.f, 0.f, 0.f, 0.f};
#pragma unroll
    for (int k0 = 0; k0 < 128; k0 += 32) {
        v16bf a = load_a_frag(arow, lane, k0);
        v16bf b = load_a_frag(WT + (n0 + (lane & 15)) * 128, lane, k0);
        acc = wmma_bf16(a, b, acc);
    }

    const int col = lane & 15;
    const int rb = (lane < 16) ? 0 : 8;
#pragma unroll
    for (int r = 0; r < 8; ++r) {
        long row = m0 + rb + r;
        float v = acc[r] + bias[n0 + col];
        if (MODE == 0) {
            Out[row * 128 + n0 + col] = (__bf16)v;
        } else {
            int m_ = (int)(row & 63);
            int s_ = (int)((row >> 6) & 31);
            int b_ = (int)(row >> 11);
            int h_ = (n0 + col) >> 5;
            int d_ = (n0 + col) & 31;
            long base = ((long)((b_ * NN + m_) * HH + h_)) * (DD * SS);
            if (MODE == 1) Out[base + s_ * DD + d_] = (__bf16)v;   // [s][d]
            else           Out[base + d_ * SS + s_] = (__bf16)v;   // [d][s]
        }
    }
}

// ---------------------------------------------------------------------------
// Kernel 2: attention stage 1. One block per (b, h, m); 128 threads = 4 waves.
// kt2/vt2 layouts make every B fragment two contiguous 8-element runs per lane.
// ctx1 stored bf16 with rows (n*N + m)*B + b, cols h*32+d.
// ---------------------------------------------------------------------------
__global__ void attn1_kernel(const __bf16* __restrict__ qt,
                             const __bf16* __restrict__ kt2,
                             const __bf16* __restrict__ vt2,
                             __bf16* __restrict__ ctx1) {
    __shared__ __bf16 sP[4][16][32];

    const int m = blockIdx.x & 63;
    const int h = (blockIdx.x >> 6) & 3;
    const int b = blockIdx.x >> 8;
    const int wave = threadIdx.x >> 5, lane = threadIdx.x & 31;
    const int n0 = wave * 16;
    const float scale = 0.17677669529663687f;   // 1/sqrt(D)

    // Q fragment (A): row = n, K = d (contiguous in qt)
    const __bf16* qrow = qt + ((long)(b * NN + n0 + (lane & 15))) * 128 + h * DD;
    v16bf aq = load_a_frag(qrow, lane, 0);

    const long hbase = ((long)((b * NN + m) * HH + h)) * (DD * SS);
    const __bf16* kb = kt2 + hbase;   // [s][d]: lane = s
    const __bf16* vb = vt2 + hbase;   // [d][s]: lane = d

    // scores: two s-tiles, contraction over d (K=32)
    v8f c0 = {0.f, 0.f, 0.f, 0.f, 0.f, 0.f, 0.f, 0.f};
    v8f c1 = c0;
    c0 = wmma_bf16(aq, load_a_frag(kb + (lane & 15) * DD, lane, 0),  c0);
    c1 = wmma_bf16(aq, load_a_frag(kb + (16 + (lane & 15)) * DD, lane, 0), c1);

    // softmax over s=32 per row (rows live in 16-lane half-groups; xor<16 stays inside)
#pragma unroll
    for (int r = 0; r < 8; ++r) {
        float s0 = c0[r] * scale, s1 = c1[r] * scale;
        float mx = fmaxf(s0, s1);
#pragma unroll
        for (int off = 1; off < 16; off <<= 1) mx = fmaxf(mx, __shfl_xor(mx, off, 32));
        float e0 = __expf(s0 - mx), e1 = __expf(s1 - mx);
        float sum = e0 + e1;
#pragma unroll
        for (int off = 1; off < 16; off <<= 1) sum += __shfl_xor(sum, off, 32);
        float inv = __frcp_rn(sum);
        c0[r] = e0 * inv;
        c1[r] = e1 * inv;
    }

    // P back through LDS into an A-fragment layout
    {
        int prow = (lane < 16) ? 0 : 8;
        int pcol = lane & 15;
#pragma unroll
        for (int r = 0; r < 8; ++r) {
            sP[wave][prow + r][pcol]      = (__bf16)c0[r];
            sP[wave][prow + r][16 + pcol] = (__bf16)c1[r];
        }
    }
    __syncthreads();

    v16bf pa = load_a_frag(&sP[wave][lane & 15][0], lane, 0);

    // ctx: contraction over s (K=32, lane = d); two d-tiles
    const int col = lane & 15;
    const int rb = (lane < 16) ? 0 : 8;
#pragma unroll
    for (int dt = 0; dt < 2; ++dt) {
        int d0 = dt * 16;
        v16bf bvf = load_a_frag(vb + (d0 + (lane & 15)) * SS, lane, 0);
        v8f acc = {0.f, 0.f, 0.f, 0.f, 0.f, 0.f, 0.f, 0.f};
        acc = wmma_bf16(pa, bvf, acc);
#pragma unroll
        for (int r = 0; r < 8; ++r) {
            int nq = n0 + rb + r;
            ctx1[(((long)(nq * NN + m)) * BB + b) * 128 + h * DD + d0 + col] = (__bf16)acc[r];
        }
    }
}

// ---------------------------------------------------------------------------
// Kernel 3: fused T = ctx1 @ t_Wo^T + bo ; ks = T @ s_Wk^T + bk ;
//                                          vs = T @ s_Wv^T + bv.
// A-tile (16x128 bf16 = 4 KB, contiguous) staged via the Tensor Data Mover;
// T tile lives only in LDS (saves 268 MB of HBM round-trip).
// ---------------------------------------------------------------------------
__global__ void fused_t_ks_vs_kernel(const __bf16* __restrict__ ctx1,
                                     const __bf16* __restrict__ WoT, const float* __restrict__ bo,
                                     const __bf16* __restrict__ WkT, const float* __restrict__ bk,
                                     const __bf16* __restrict__ WvT, const float* __restrict__ bv,
                                     __bf16* __restrict__ ks, __bf16* __restrict__ vs) {
    __shared__ __bf16 sA[16 * 128];
    __shared__ __bf16 sT[16 * 128];
    const int tid = threadIdx.x;
    const long m0 = (long)blockIdx.x * 16;

    if (tid < 32) tdm_load_tile_4kb(ctx1 + m0 * 128, sA);
    __syncthreads();

    const int wave = tid >> 5, lane = tid & 31;
    const int n0 = wave * 16;
    const int col = lane & 15;
    const int rb = (lane < 16) ? 0 : 8;
    const __bf16* arow = sA + (lane & 15) * 128;
    const __bf16* trow = sT + (lane & 15) * 128;

    // stage 1: T tile
    v8f acc = {0.f, 0.f, 0.f, 0.f, 0.f, 0.f, 0.f, 0.f};
#pragma unroll
    for (int k0 = 0; k0 < 128; k0 += 32) {
        v16bf a = load_a_frag(arow, lane, k0);
        v16bf b = load_a_frag(WoT + (n0 + (lane & 15)) * 128, lane, k0);
        acc = wmma_bf16(a, b, acc);
    }
#pragma unroll
    for (int r = 0; r < 8; ++r)
        sT[(rb + r) * 128 + n0 + col] = (__bf16)(acc[r] + bo[n0 + col]);
    __syncthreads();

    // stage 2: ks & vs tiles from LDS-resident T
    v8f ak = {0.f, 0.f, 0.f, 0.f, 0.f, 0.f, 0.f, 0.f};
    v8f av = ak;
#pragma unroll
    for (int k0 = 0; k0 < 128; k0 += 32) {
        v16bf a = load_a_frag(trow, lane, k0);
        v16bf bK = load_a_frag(WkT + (n0 + (lane & 15)) * 128, lane, k0);
        v16bf bV = load_a_frag(WvT + (n0 + (lane & 15)) * 128, lane, k0);
        ak = wmma_bf16(a, bK, ak);
        av = wmma_bf16(a, bV, av);
    }
#pragma unroll
    for (int r = 0; r < 8; ++r) {
        long row = m0 + rb + r;
        ks[row * 128 + n0 + col] = (__bf16)(ak[r] + bk[n0 + col]);
        vs[row * 128 + n0 + col] = (__bf16)(av[r] + bv[n0 + col]);
    }
}

// ---------------------------------------------------------------------------
// Kernel 4: attention stage 2 (masked, over m=64). One block per (n, b);
// 256 threads as (h,m) pairs. ~134 MFLOP total -> VALU is fine.
// ---------------------------------------------------------------------------
__global__ void attn2_kernel(const __bf16* __restrict__ qs,
                             const __bf16* __restrict__ ks,
                             const __bf16* __restrict__ vs,
                             const int* __restrict__ adj,
                             __bf16* __restrict__ ctx2) {
    __shared__ float sS[HH][NN];
    const int n = blockIdx.x >> 6;
    const int b = blockIdx.x & 63;
    const int tid = threadIdx.x;
    const int h = tid >> 6, m = tid & 63;
    const float scale = 0.17677669529663687f;

    const __bf16* qrow = qs + ((long)(b * NN + n)) * 128 + h * DD;
    const __bf16* krow = ks + (((long)(n * NN + m)) * BB + b) * 128 + h * DD;
    v16bf q0 = load_v16bf(qrow), q1 = load_v16bf(qrow + 16);
    v16bf k0 = load_v16bf(krow), k1 = load_v16bf(krow + 16);
    float s = 0.f;
#pragma unroll
    for (int e = 0; e < 16; ++e) s += (float)q0[e] * (float)k0[e];
#pragma unroll
    for (int e = 0; e < 16; ++e) s += (float)q1[e] * (float)k1[e];
    s *= scale;
    if (adj[n * NN + m] <= 0) s = -1.0e9f;
    sS[h][m] = s;
    __syncthreads();

    float mx = -3.4e38f;
    for (int j = 0; j < NN; ++j) mx = fmaxf(mx, sS[h][j]);
    __syncthreads();
    float e = __expf(s - mx);
    sS[h][m] = e;
    __syncthreads();
    float sum = 0.f;
    for (int j = 0; j < NN; ++j) sum += sS[h][j];
    float p = e * __frcp_rn(sum);
    __syncthreads();
    sS[h][m] = p;
    __syncthreads();

    if (tid < 128) {
        int oh = tid >> 5, od = tid & 31;
        float acc = 0.f;
        for (int j = 0; j < NN; ++j)
            acc += sS[oh][j] *
                   (float)vs[(((long)(n * NN + j)) * BB + b) * 128 + oh * DD + od];
        ctx2[((long)(n * BB + b)) * 128 + oh * DD + od] = (__bf16)acc;
    }
}

// ---------------------------------------------------------------------------
// Kernel 5: out = ctx2 @ s_Wo^T + s_bo, fp32, with (n,b) -> (b,n) transpose.
// A tile staged via TDM as well.
// ---------------------------------------------------------------------------
__global__ void out_gemm_kernel(const __bf16* __restrict__ ctx2,
                                const __bf16* __restrict__ WoT,
                                const float* __restrict__ bo,
                                float* __restrict__ out) {
    __shared__ __bf16 sA[16 * 128];
    const int tid = threadIdx.x;
    const long m0 = (long)blockIdx.x * 16;

    if (tid < 32) tdm_load_tile_4kb(ctx2 + m0 * 128, sA);
    __syncthreads();

    const int wave = tid >> 5, lane = tid & 31;
    const int n0 = wave * 16;
    const __bf16* arow = sA + (lane & 15) * 128;

    v8f acc = {0.f, 0.f, 0.f, 0.f, 0.f, 0.f, 0.f, 0.f};
#pragma unroll
    for (int k0 = 0; k0 < 128; k0 += 32) {
        v16bf a = load_a_frag(arow, lane, k0);
        v16bf b = load_a_frag(WoT + (n0 + (lane & 15)) * 128, lane, k0);
        acc = wmma_bf16(a, b, acc);
    }

    const int col = lane & 15;
    const int rb = (lane < 16) ? 0 : 8;
#pragma unroll
    for (int r = 0; r < 8; ++r) {
        long rsrc = m0 + rb + r;        // rsrc = n*B + b
        int nn = (int)(rsrc >> 6);
        int bb = (int)(rsrc & 63);
        out[((long)(bb * NN + nn)) * 128 + n0 + col] = acc[r] + bo[n0 + col];
    }
}

// ---------------------------------------------------------------------------
extern "C" void kernel_launch(void* const* d_in, const int* in_sizes, int n_in,
                              void* d_out, int out_size, void* d_ws, size_t ws_size,
                              hipStream_t stream) {
    const float* x    = (const float*)d_in[0];
    const int*   adj  = (const int*)d_in[1];
    const float* t_Wq = (const float*)d_in[2];
    const float* t_bq = (const float*)d_in[3];
    const float* t_Wk = (const float*)d_in[4];
    const float* t_bk = (const float*)d_in[5];
    const float* t_Wv = (const float*)d_in[6];
    const float* t_bv = (const float*)d_in[7];
    const float* t_Wo = (const float*)d_in[8];
    const float* t_bo = (const float*)d_in[9];
    const float* s_Wq = (const float*)d_in[10];
    const float* s_bq = (const float*)d_in[11];
    const float* s_Wk = (const float*)d_in[12];
    const float* s_bk = (const float*)d_in[13];
    const float* s_Wv = (const float*)d_in[14];
    const float* s_bv = (const float*)d_in[15];
    const float* s_Wo = (const float*)d_in[16];
    const float* s_bo = (const float*)d_in[17];
    float* out = (float*)d_out;

    // Workspace layout (bf16 intermediates)
    char* w = (char*)d_ws;
    __bf16* wb     = (__bf16*)w;                       // 8 x 16384 bf16 (row-major W[n][k])
    __bf16* wb_tWq = wb + 0 * 16384;
    __bf16* wb_tWk = wb + 1 * 16384;
    __bf16* wb_tWv = wb + 2 * 16384;
    __bf16* wb_tWo = wb + 3 * 16384;
    __bf16* wb_sWq = wb + 4 * 16384;
    __bf16* wb_sWk = wb + 5 * 16384;
    __bf16* wb_sWv = wb + 6 * 16384;
    __bf16* wb_sWo = wb + 7 * 16384;
    size_t off = 8 * 16384 * sizeof(__bf16);

    __bf16* qt   = (__bf16*)(w + off); off += (size_t)BB * NN * EE * 2;           // 1 MB
    __bf16* qs   = (__bf16*)(w + off); off += (size_t)BB * NN * EE * 2;           // 1 MB
    __bf16* kt2  = (__bf16*)(w + off); off += (size_t)BB * SS * NN * EE * 2;      // 33.5 MB
    __bf16* vt2  = (__bf16*)(w + off); off += (size_t)BB * SS * NN * EE * 2;      // 33.5 MB
    __bf16* ctx1 = (__bf16*)(w + off); off += (size_t)NN * NN * BB * EE * 2;      // 67 MB
    __bf16* ksb  = (__bf16*)(w + off); off += (size_t)NN * NN * BB * EE * 2;      // 67 MB
    __bf16* vsb  = (__bf16*)(w + off); off += (size_t)NN * NN * BB * EE * 2;      // 67 MB
    __bf16* ctx2 = (__bf16*)(w + off); off += (size_t)NN * BB * EE * 2;           // 1 MB
    (void)ws_size; (void)in_sizes; (void)n_in; (void)out_size;

    // 0) weights -> bf16 (same W[n][k] layout)
    wconv_kernel<<<(8 * 16384) / 256, 256, 0, stream>>>(
        t_Wq, t_Wk, t_Wv, t_Wo, s_Wq, s_Wk, s_Wv, s_Wo, wb);

    // 1) projections (innerShift: 6 for the q remap, 17 = log2(B*S*N) contiguous)
    const float* qbase = x + (size_t)(SS - 1) * NN * EE;   // x[:, S-1]
    proj_gemm_kernel<0><<<(BB * NN) / 16, 256, 0, stream>>>(
        qbase, wb_tWq, t_bq, qt, 6, (long)SS * NN * EE);
    proj_gemm_kernel<0><<<(BB * NN) / 16, 256, 0, stream>>>(
        qbase, wb_sWq, s_bq, qs, 6, (long)SS * NN * EE);
    proj_gemm_kernel<1><<<(BB * SS * NN) / 16, 256, 0, stream>>>(
        x, wb_tWk, t_bk, kt2, 17, 0L);
    proj_gemm_kernel<2><<<(BB * SS * NN) / 16, 256, 0, stream>>>(
        x, wb_tWv, t_bv, vt2, 17, 0L);

    // 2) attention stage 1  -> ctx1
    attn1_kernel<<<BB * HH * NN, 128, 0, stream>>>(qt, kt2, vt2, ctx1);

    // 3) fused T -> ks, vs  (TDM-staged A tiles)
    fused_t_ks_vs_kernel<<<(NN * NN * BB) / 16, 256, 0, stream>>>(
        ctx1, wb_tWo, t_bo, wb_sWk, s_bk, wb_sWv, s_bv, ksb, vsb);

    // 4) attention stage 2 (masked) -> ctx2
    attn2_kernel<<<NN * BB, 256, 0, stream>>>(qs, ksb, vsb, adj, ctx2);

    // 5) output GEMM with transpose (TDM-staged A tile)
    out_gemm_kernel<<<(NN * BB) / 16, 256, 0, stream>>>(ctx2, wb_sWo, s_bo, out);
}
